// MoECNN_7919919694088
// MI455X (gfx1250) — matrix operand order repaired and checked by reading, not compiled, not run on hardware
//
#include <hip/hip_runtime.h>

typedef __attribute__((ext_vector_type(16))) __bf16 bfvec16;
typedef __attribute__((ext_vector_type(8)))  float  fvec8;

#define NB   64
#define NE   8
#define HH   124
#define WW   124
#define HWSZ (HH*WW)        // 15376
#define C1   32
#define C2   64
#define PH   62
#define PWD  62
#define NPIX (PH*PWD)       // 3844
#define FCIN (C2*NPIX)      // 246016
#define KDIM (C1*9)         // 288
#define BN_N ((float)(NB*HH*WW))

// ---------------------------------------------------------------- router
__global__ void k_router(const float* __restrict__ x,
                         const float* __restrict__ rw,
                         const float* __restrict__ rb,
                         float* __restrict__ gates) {
  int b = blockIdx.x;
  int t = threadIdx.x;
  __shared__ float red[256];
  __shared__ float pooled[3];
  for (int c = 0; c < 3; ++c) {
    const float* px = x + ((size_t)(b*3 + c)) * HWSZ;
    float s = 0.f;
    for (int i = t; i < HWSZ; i += 256) s += px[i];
    red[t] = s;
    __syncthreads();
    for (int off = 128; off > 0; off >>= 1) {
      if (t < off) red[t] += red[t + off];
      __syncthreads();
    }
    if (t == 0) pooled[c] = red[0] / (float)HWSZ;
    __syncthreads();
  }
  if (t == 0) {
    float lg[NE];
    for (int e = 0; e < NE; ++e)
      lg[e] = pooled[0]*rw[e*3+0] + pooled[1]*rw[e*3+1] + pooled[2]*rw[e*3+2] + rb[e];
    int i0 = 0;
    for (int e = 1; e < NE; ++e) if (lg[e] > lg[i0]) i0 = e;
    int i1 = (i0 == 0) ? 1 : 0;
    for (int e = 0; e < NE; ++e) if (e != i0 && lg[e] > lg[i1]) i1 = e;
    float g1 = __expf(lg[i1] - lg[i0]);      // softmax over top-2
    float inv = 1.f / (1.f + g1);
    for (int e = 0; e < NE; ++e) gates[b*NE + e] = 0.f;
    gates[b*NE + i0] = inv;
    gates[b*NE + i1] = g1 * inv;
  }
}

// ----------------------------------------------------- workspace zero + out init
__global__ void k_zero_stats(float* gsum, float* gssq) {
  int t = blockIdx.x * blockDim.x + threadIdx.x;
  if (t < NE*C1) { gsum[t] = 0.f; gssq[t] = 0.f; }
}

__global__ void k_out_init(const float* __restrict__ gates,
                           const float* __restrict__ fcb,
                           float* __restrict__ out) {
  int i = blockIdx.x * blockDim.x + threadIdx.x;
  if (i < NB*10) {
    int b = i / 10, d = i % 10;
    float s = 0.f;
    for (int e = 0; e < NE; ++e) s += gates[b*NE + e] * fcb[e*10 + d];
    out[i] = s;
  }
}

__global__ void k_bn_finalize(const float* __restrict__ gsum,
                              const float* __restrict__ gssq,
                              float* __restrict__ gmu,
                              float* __restrict__ grstd) {
  int i = blockIdx.x * blockDim.x + threadIdx.x;
  if (i < NE*C1) {
    float mu  = gsum[i] / BN_N;
    float var = gssq[i] / BN_N - mu*mu;          // biased variance (matches ref)
    gmu[i]   = mu;
    grstd[i] = rsqrtf(var + 1e-5f);
  }
}

// ================ conv1 as WMMA implicit GEMM (K = 27 padded to 32) ==========
// A = conv1 weights [32 oc][32 k'], k' = kw*9 + ic*3 + kh (27..31 zero).
// B = input pixels: 3 phase-shifted slabs (one per kw) so every per-lane
// fragment is a 16B-aligned contiguous 16-element run at base pxb.
// One block = one conv-row pair (y0 = 2*z, y0+1) -> feeds one 2x2 pool row.
#define INSW1 128                          // slab row stride (256 B)
#define SROWS1 36                          // [ph 3][ic 3][iy 4]
#define ZROW  (SROWS1*INSW1)               // zero row for k >= 27
#define SLAB1 (ZROW + INSW1)               // 4736 elements

// per-lane B-row decode, shared by both conv1 kernels
__device__ inline void conv1_rowoff(int kin, int& r0off, int& r1off) {
  if (kin < 27) {
    int kw = kin / 9, rem = kin % 9, ic = rem / 3, kh = rem % 3;
    int br = (kw*3 + ic)*4 + kh;
    r0off = br * INSW1;
    r1off = (br + 1) * INSW1;
  } else {
    r0off = ZROW; r1off = ZROW;
  }
}

__device__ inline void conv1_stage(int t, int e, int b, int y0,
                                   const float* __restrict__ x,
                                   const float* __restrict__ w1,
                                   __bf16* wA1, __bf16* slab) {
  for (int i = t; i < C1*32; i += 128) {
    int o = i >> 5, kp = i & 31;
    __bf16 v = (__bf16)0.f;
    if (kp < 27) {
      int kw = kp / 9, rem = kp % 9, ic = rem / 3, kh = rem % 3;
      v = (__bf16)w1[(size_t)e*C1*27 + o*27 + ic*9 + kh*3 + kw];
    }
    wA1[i] = v;
  }
  for (int i = t; i < SLAB1; i += 128) {
    int row = i / INSW1, j = i % INSW1;
    __bf16 v = (__bf16)0.f;
    if (row < SROWS1) {
      int ph = row / 12;
      int r2 = row % 12;
      int ic = r2 / 4, iy = r2 % 4;
      int gy = y0 - 1 + iy;
      int gx = j - 1 + ph;
      if (gy >= 0 && gy < HH && gx >= 0 && gx < WW)
        v = (__bf16)x[((size_t)(b*3 + ic)) * HWSZ + gy*WW + gx];
    }
    slab[i] = v;
  }
}

// -------- dense pass: conv1 + bias -> BN sum / sumsq (cannot be gated) -------
__global__ void __launch_bounds__(128)
k_conv1_wmma_stats(const float* __restrict__ x,
                   const float* __restrict__ w1,
                   const float* __restrict__ b1,
                   float* __restrict__ gsum,
                   float* __restrict__ gssq) {
  int e = blockIdx.x, b = blockIdx.y, y0 = 2*blockIdx.z;
  int t = threadIdx.x;
  int lane = t & 31, wv = t >> 5;
  int hi = (lane >= 16) ? 1 : 0;
  int n  = lane & 15;

  __shared__ alignas(16) __bf16 wA1[C1*32];    // 2048 B
  __shared__ alignas(16) __bf16 slab[SLAB1];   // 9472 B
  __shared__ float b1s[C1];
  __shared__ float ssum[C1], ssq[C1];

  conv1_stage(t, e, b, y0, x, w1, wA1, slab);
  if (t < C1) { b1s[t] = b1[e*C1 + t]; ssum[t] = 0.f; ssq[t] = 0.f; }
  __syncthreads();

  int kin = n + hi*16;
  int r0off, r1off;
  conv1_rowoff(kin, r0off, r1off);

  // A fragments: loaded once, reused by every tile (single k-step)
  bfvec16 av0, av1;
  {
    const __bf16* a0 = wA1 + n*32 + hi*8;
    const __bf16* a1 = wA1 + (16 + n)*32 + hi*8;
    #pragma unroll
    for (int j = 0; j < 8; ++j) {
      av0[j] = a0[j];  av0[8+j] = a0[16+j];
      av1[j] = a1[j];  av1[8+j] = a1[16+j];
    }
  }

  float csum[16], csq[16];
  #pragma unroll
  for (int i = 0; i < 16; ++i) { csum[i] = 0.f; csq[i] = 0.f; }

  #pragma unroll
  for (int ti = 0; ti < 2; ++ti) {          // 8 N-tiles / 4 waves
    int pxb = (wv*2 + ti) * 16;
    bfvec16 bv0, bv1;
    const __bf16* bp = slab + pxb + n;
    #pragma unroll
    for (int j = 0; j < 16; ++j) {
      int kk = (j < 8) ? (hi*8 + j) : (16 + hi*8 + j - 8);
      int o0, o1; conv1_rowoff(kk, o0, o1);
      bv0[j] = bp[o0]; bv1[j] = bp[o1];
    }
    fvec8 aY0O0 = {}, aY0O1 = {}, aY1O0 = {}, aY1O1 = {};
    aY0O0 = __builtin_amdgcn_wmma_f32_16x16x32_bf16(false, av0, false, bv0, (short)0, aY0O0, false, false);
    aY0O1 = __builtin_amdgcn_wmma_f32_16x16x32_bf16(false, av1, false, bv0, (short)0, aY0O1, false, false);
    aY1O0 = __builtin_amdgcn_wmma_f32_16x16x32_bf16(false, av0, false, bv1, (short)0, aY1O0, false, false);
    aY1O1 = __builtin_amdgcn_wmma_f32_16x16x32_bf16(false, av1, false, bv1, (short)0, aY1O1, false, false);
    bool valid = (pxb + n) < WW;
    #pragma unroll
    for (int ot = 0; ot < 2; ++ot) {
      #pragma unroll
      for (int v = 0; v < 8; ++v) {
        int o = ot*16 + v + hi*8;
        float c0 = ((ot == 0) ? aY0O0[v] : aY0O1[v]) + b1s[o];
        float c1 = ((ot == 0) ? aY1O0[v] : aY1O1[v]) + b1s[o];
        if (!valid) { c0 = 0.f; c1 = 0.f; }
        csum[ot*8+v] += c0 + c1;
        csq[ot*8+v]  += c0*c0 + c1*c1;
      }
    }
  }
  // cross-lane reduce over the 16 N lanes, then one LDS atomic per channel
  #pragma unroll
  for (int i = 0; i < 16; ++i) {
    float r = csum[i], q = csq[i];
    #pragma unroll
    for (int off = 1; off <= 8; off <<= 1) {
      r += __shfl_xor(r, off);
      q += __shfl_xor(q, off);
    }
    if (n == 0) {
      int o = (i >> 3)*16 + (i & 7) + hi*8;
      atomicAdd(&ssum[o], r);
      atomicAdd(&ssq[o],  q);
    }
  }
  __syncthreads();
  if (t < C1) {
    atomicAdd(&gsum[e*C1 + t], ssum[t]);
    atomicAdd(&gssq[e*C1 + t], ssq[t]);
  }
}

// -------- gated pass: conv1 + BN + ReLU + 2x2 maxpool -> bf16 pooled ---------
__global__ void __launch_bounds__(128)
k_conv1_wmma_pool(const float* __restrict__ x,
                  const float* __restrict__ w1,
                  const float* __restrict__ b1,
                  const float* __restrict__ bng,
                  const float* __restrict__ bnb,
                  const float* __restrict__ gmu,
                  const float* __restrict__ grstd,
                  const float* __restrict__ gates,
                  __bf16* __restrict__ pooled) {
  int e = blockIdx.x, b = blockIdx.y, py = blockIdx.z;      // one pooled row
  if (gates[b*NE + e] == 0.f) return;                       // top-2 sparsity
  int y0 = 2*py;
  int t = threadIdx.x;
  int lane = t & 31, wv = t >> 5;
  int hi = (lane >= 16) ? 1 : 0;
  int n  = lane & 15;

  __shared__ alignas(16) __bf16 wA1[C1*32];
  __shared__ alignas(16) __bf16 slab[SLAB1];
  __shared__ float b1s[C1], sc[C1], sh[C1];

  conv1_stage(t, e, b, y0, x, w1, wA1, slab);
  if (t < C1) {
    b1s[t] = b1[e*C1 + t];
    float s = bng[e*C1 + t] * grstd[e*C1 + t];
    sc[t] = s;
    sh[t] = bnb[e*C1 + t] - gmu[e*C1 + t] * s;
  }
  __syncthreads();

  int kin = n + hi*16;
  int r0off, r1off;
  conv1_rowoff(kin, r0off, r1off);

  bfvec16 av0, av1;
  {
    const __bf16* a0 = wA1 + n*32 + hi*8;
    const __bf16* a1 = wA1 + (16 + n)*32 + hi*8;
    #pragma unroll
    for (int j = 0; j < 8; ++j) {
      av0[j] = a0[j];  av0[8+j] = a0[16+j];
      av1[j] = a1[j];  av1[8+j] = a1[16+j];
    }
  }

  size_t pbase = ((size_t)(e*NB + b)) * C1 * NPIX;
  #pragma unroll
  for (int ti = 0; ti < 2; ++ti) {
    int pxb = (wv*2 + ti) * 16;
    bfvec16 bv0, bv1;
    const __bf16* bp = slab + pxb + n;
    #pragma unroll
    for (int j = 0; j < 16; ++j) {
      int kk = (j < 8) ? (hi*8 + j) : (16 + hi*8 + j - 8);
      int o0, o1; conv1_rowoff(kk, o0, o1);
      bv0[j] = bp[o0]; bv1[j] = bp[o1];
    }
    fvec8 aY0O0 = {}, aY0O1 = {}, aY1O0 = {}, aY1O1 = {};
    aY0O0 = __builtin_amdgcn_wmma_f32_16x16x32_bf16(false, av0, false, bv0, (short)0, aY0O0, false, false);
    aY0O1 = __builtin_amdgcn_wmma_f32_16x16x32_bf16(false, av1, false, bv0, (short)0, aY0O1, false, false);
    aY1O0 = __builtin_amdgcn_wmma_f32_16x16x32_bf16(false, av0, false, bv1, (short)0, aY1O0, false, false);
    aY1O1 = __builtin_amdgcn_wmma_f32_16x16x32_bf16(false, av1, false, bv1, (short)0, aY1O1, false, false);
    int px = pxb + n;
    #pragma unroll
    for (int ot = 0; ot < 2; ++ot) {
      #pragma unroll
      for (int v = 0; v < 8; ++v) {
        int o = ot*16 + v + hi*8;
        float t0 = (((ot == 0) ? aY0O0[v] : aY0O1[v]) + b1s[o]) * sc[o] + sh[o];
        float t1 = (((ot == 0) ? aY1O0[v] : aY1O1[v]) + b1s[o]) * sc[o] + sh[o];
        float m  = fmaxf(t0, t1);                    // vertical pool
        float m2 = fmaxf(m, __shfl_xor(m, 1));       // horizontal pool (lane pair)
        if ((n & 1) == 0 && px < WW) {
          int pp = px >> 1;                          // pooled column 0..61
          pooled[pbase + (size_t)o * NPIX + py*PWD + pp] = (__bf16)fmaxf(m2, 0.f);
        }
      }
    }
  }
}

// --------------- gated conv2 (implicit GEMM, bf16 WMMA) fused with FC epilogue
#define INSW   72                     // row stride (144 B, 16B aligned)
#define PH3    (C1*3*INSW)            // one phase: 6912 elements
__global__ void __launch_bounds__(128)
k_conv2_fc(const __bf16* __restrict__ pooled,
           const float* __restrict__ w2,
           const float* __restrict__ cb2,
           const float* __restrict__ fcw,
           const float* __restrict__ gates,
           float* __restrict__ out) {
  int e = blockIdx.x, b = blockIdx.y, r0 = blockIdx.z;      // one conv2 row / block
  float gate = gates[b*NE + e];
  if (gate == 0.f) return;                                  // top-2 sparsity: skip
  int t = threadIdx.x;
  int lane = t & 31, wv = t >> 5;                           // 4 waves = 4 pixel tiles
  int hi = (lane >= 16) ? 1 : 0;
  int n  = lane & 15;

  __shared__ alignas(16) __bf16 wA[C2 * KDIM];  // [o][k'] 36864 B
  __shared__ alignas(16) __bf16 ins[2 * PH3];   // dual-phase slab 27648 B
  __shared__ float  cb2s[C2];
  __shared__ float  fcsh[16];

  // stage weights reordered to k' = kw*96 + ic*3 + kh
  const float* w2e = w2 + (size_t)e * C2 * KDIM;
  for (int i = t; i < C2*KDIM; i += 128) {
    int o  = i / KDIM, kp = i % KDIM;
    int kw = kp / 96, rem = kp % 96;
    int ic = rem / 3, kh = rem % 3;
    wA[i] = (__bf16)w2e[o*KDIM + ic*9 + kh*3 + kw];
  }
  // dual-phase input slab, rows r0-1..r0+1
  size_t pbase = ((size_t)(e*NB + b)) * C1 * NPIX;
  for (int i = t; i < 2*PH3; i += 128) {
    int ph  = i / PH3;
    int rem = i % PH3;
    int c  = rem / (3*INSW);
    int r2 = rem % (3*INSW);
    int kh = r2 / INSW, j = r2 % INSW;
    int gy = r0 + kh - 1;
    int gx = j - 1 + ph;
    __bf16 v = (__bf16)0.f;
    if (gy >= 0 && gy < PH && gx >= 0 && gx < PWD)
      v = pooled[pbase + (size_t)c * NPIX + gy*PWD + gx];
    ins[i] = v;
  }
  if (t < C2) cb2s[t] = cb2[e*C2 + t];
  if (t < 16) fcsh[t] = 0.f;
  __syncthreads();

  int kin = n + hi*16;
  int rowb[3];
  #pragma unroll
  for (int c3 = 0; c3 < 3; ++c3) {
    int rem = c3*32 + kin;
    int ic = rem / 3, kh = rem % 3;
    rowb[c3] = (ic*3 + kh) * INSW;
  }
  int anb = n * KDIM + hi*8;

  int pxb = wv * 16;
  fvec8 acc0 = {}, acc1 = {}, acc2 = {}, acc3 = {};

  #pragma unroll
  for (int ks = 0; ks < 9; ++ks) {
    int kw = ks / 3;                        // compile-time after unroll
    const __bf16* bp = ins + ((kw == 1) ? PH3 : 0)
                           + pxb + n + ((kw == 2) ? 2 : 0);
    bfvec16 bv;
    #pragma unroll
    for (int j = 0; j < 16; ++j) {
      int kk = (j < 8) ? (hi*8 + j) : (16 + hi*8 + j - 8);
      int rm = (ks % 3)*32 + kk;
      bv[j] = bp[((rm / 3)*3 + rm % 3) * INSW];
    }
    #pragma unroll
    for (int ot = 0; ot < 4; ++ot) {
      const __bf16* ap = wA + anb + ot*16*KDIM + ks*32;
      bfvec16 av;
      #pragma unroll
      for (int j = 0; j < 8; ++j) { av[j] = ap[j]; av[8+j] = ap[16+j]; }
      fvec8& ac = (ot == 0) ? acc0 : (ot == 1) ? acc1 : (ot == 2) ? acc2 : acc3;
      ac = __builtin_amdgcn_wmma_f32_16x16x32_bf16(
               false, av, false, bv, (short)0, ac, false, false);
    }
  }

  // epilogue: D layout -> N = lane&15 = pixel, M = vgpr + 8*hi = channel-in-tile
  float fcacc[10];
  #pragma unroll
  for (int d = 0; d < 10; ++d) fcacc[d] = 0.f;
  int px = pxb + n;
  if (px < PWD) {
    int gp = r0 * PWD + px;
    const float* fwb = fcw + ((size_t)(e*10)) * FCIN + gp;
    __builtin_prefetch(fwb, 0, 1);
    #pragma unroll
    for (int ot = 0; ot < 4; ++ot) {
      fvec8 ac = (ot == 0) ? acc0 : (ot == 1) ? acc1 : (ot == 2) ? acc2 : acc3;
      #pragma unroll
      for (int v = 0; v < 8; ++v) {
        int o = ot*16 + v + hi*8;
        float val = fmaxf(ac[v] + cb2s[o], 0.f);
        const float* fw = fwb + (size_t)o * NPIX;
        #pragma unroll
        for (int d = 0; d < 10; ++d)
          fcacc[d] = fmaf(val, fw[(size_t)d * FCIN], fcacc[d]);
      }
    }
  }
  #pragma unroll
  for (int d = 0; d < 10; ++d) atomicAdd(&fcsh[d], fcacc[d]);
  __syncthreads();
  if (t < 10) atomicAdd(&out[b*10 + t], gate * fcsh[t]);
}

// ---------------------------------------------------------------- launcher
extern "C" void kernel_launch(void* const* d_in, const int* in_sizes, int n_in,
                              void* d_out, int out_size, void* d_ws, size_t ws_size,
                              hipStream_t stream) {
  (void)in_sizes; (void)n_in; (void)out_size; (void)ws_size;
  const float* x   = (const float*)d_in[0];
  const float* w1  = (const float*)d_in[1];
  const float* b1  = (const float*)d_in[2];
  const float* bng = (const float*)d_in[3];
  const float* bnb = (const float*)d_in[4];
  const float* w2  = (const float*)d_in[5];
  const float* cb2 = (const float*)d_in[6];
  const float* fcw = (const float*)d_in[7];
  const float* fcb = (const float*)d_in[8];
  const float* rw  = (const float*)d_in[9];
  const float* rb  = (const float*)d_in[10];
  float* out = (float*)d_out;

  float* gates = (float*)d_ws;            // 512 f
  float* gsum  = gates + NB*NE;           // 256 f
  float* gssq  = gsum  + NE*C1;           // 256 f
  float* gmu   = gssq  + NE*C1;           // 256 f
  float* grstd = gmu   + NE*C1;           // 256 f
  __bf16* pooled = (__bf16*)(grstd + NE*C1);   // E*B*32*3844 bf16 (~126 MB)

  k_router<<<NB, 256, 0, stream>>>(x, rw, rb, gates);
  k_zero_stats<<<1, 256, 0, stream>>>(gsum, gssq);
  k_out_init<<<3, 256, 0, stream>>>(gates, fcb, out);
  k_conv1_wmma_stats<<<dim3(NE, NB, PH), 128, 0, stream>>>(x, w1, b1, gsum, gssq);
  k_bn_finalize<<<1, 256, 0, stream>>>(gsum, gssq, gmu, grstd);
  k_conv1_wmma_pool<<<dim3(NE, NB, PH), 128, 0, stream>>>(
      x, w1, b1, bng, bnb, gmu, grstd, gates, pooled);
  k_conv2_fc<<<dim3(NE, NB, PH), 128, 0, stream>>>(
      pooled, w2, cb2, fcw, gates, out);
}
